// Attention_86131274154152
// MI455X (gfx1250) — matrix operand order repaired and checked by reading, not compiled, not run on hardware
//
#include <hip/hip_runtime.h>
#include <hip/hip_bf16.h>

// ---------------------------------------------------------------------------
// Masked attention block for MI455X (gfx1250, wave32, WMMA).
//   qkv = x @ W_qkv ; per-head flash softmax with post-softmax label mask
//   (mask multiplies the numerator only, denominator stays unmasked, matching
//   the reference's "no renormalization" semantics) ; out = attn_out @ W_out.
// All matmuls: v_wmma_f32_16x16x32_bf16, f32 accumulation, softmax in f32.
// ---------------------------------------------------------------------------

#define B_TOK 4096
#define DIM   768
#define HEADS 8
#define DH    96
#define INNER 768   // HEADS * DH
#define NQKV  2304  // 3 * INNER

typedef __bf16 bf16;
typedef __attribute__((ext_vector_type(8)))  __bf16 v8bf;
typedef __attribute__((ext_vector_type(16))) __bf16 v16bf;
typedef __attribute__((ext_vector_type(8)))  float  v8f;

#define WMMA_BF16(a, b, c) \
  __builtin_amdgcn_wmma_f32_16x16x32_bf16(false, (a), false, (b), (short)0, (c), false, false)

static __device__ inline v8f zero_v8f() {
  v8f z;
#pragma unroll
  for (int i = 0; i < 8; ++i) z[i] = 0.0f;
  return z;
}

// A-fragment, 16x32 bf16, row-major source (ld in elements).
// ISA layout (CDNA5 ISA "16-bit A-Matrix 16x32"): lanes 0-15 hold row M=lane,
// K=0..7 (v0..v3) and K=16..23 (v4..v7); lanes 16-31 hold K=8..15 / 24..31.
static __device__ inline v16bf load_frag_a(const bf16* __restrict__ base, int ld) {
  const int lane = threadIdx.x & 31;
  const bf16* p = base + (size_t)(lane & 15) * ld + ((lane >> 4) << 3);
  v8bf lo = *(const v8bf*)(p);
  v8bf hi = *(const v8bf*)(p + 16);
  v16bf r;
#pragma unroll
  for (int i = 0; i < 8; ++i) { r[i] = lo[i]; r[i + 8] = hi[i]; }
  return r;
}

// B-fragment, 32x16 bf16, consumed from B^T stored row-major [N][K]:
// lane holds one N-row; lanes 0-15 -> K=0..7 / 16..23, lanes 16-31 -> K=8..15 / 24..31
// (same K order as the A-fragment).
static __device__ inline v16bf load_frag_b(const bf16* __restrict__ base, int ld) {
  const int lane = threadIdx.x & 31;
  const bf16* p = base + (size_t)(lane & 15) * ld + ((lane >> 4) << 3);
  v8bf lo = *(const v8bf*)(p);
  v8bf hi = *(const v8bf*)(p + 16);
  v16bf r;
#pragma unroll
  for (int i = 0; i < 8; ++i) { r[i] = lo[i]; r[i + 8] = hi[i]; }
  return r;
}

// ---------------------------------------------------------------------------
// Prep kernels (bandwidth-trivial: ~33 MB read+written)
// ---------------------------------------------------------------------------
__global__ void cvt_f32_bf16_kernel(const float* __restrict__ src,
                                    bf16* __restrict__ dst, int n) {
  int i = blockIdx.x * blockDim.x + threadIdx.x;
  if (i < n) dst[i] = (bf16)src[i];
}

// dst[c][r] = (bf16)src[r][c]  (src is [rows][cols] f32)
__global__ void transpose_f32_bf16_kernel(const float* __restrict__ src,
                                          bf16* __restrict__ dst,
                                          int rows, int cols) {
  int i = blockIdx.x * blockDim.x + threadIdx.x;
  if (i >= rows * cols) return;
  int r = i / cols, c = i % cols;
  dst[(size_t)c * rows + r] = (bf16)src[i];
}

// ---------------------------------------------------------------------------
// QKV GEMM: [4096,768] x [768,2304] -> q,k row-major bf16; v stored transposed
// per head so the P*V matmul's B fragments are contiguous.
// 8 waves/block, wave tile 32x32, block tile 64(M) x 128(N).
// ---------------------------------------------------------------------------
static __device__ inline void store_qkv_tile(const v8f c, int mt, int nt,
                                             bf16* __restrict__ q,
                                             bf16* __restrict__ k,
                                             bf16* __restrict__ vT) {
  const int lane = threadIdx.x & 31;
  const int ncol = nt + (lane & 15);
  const int m0 = mt + ((lane >> 4) << 3);
  if (ncol < INNER) {                       // Q
#pragma unroll
    for (int r = 0; r < 8; ++r) q[(size_t)(m0 + r) * INNER + ncol] = (bf16)c[r];
  } else if (ncol < 2 * INNER) {            // K
    const int cc = ncol - INNER;
#pragma unroll
    for (int r = 0; r < 8; ++r) k[(size_t)(m0 + r) * INNER + cc] = (bf16)c[r];
  } else {                                  // V -> vT[h*96+d][b], 16B store
    const int cc = ncol - 2 * INNER;
    v8bf pk;
#pragma unroll
    for (int r = 0; r < 8; ++r) pk[r] = (bf16)c[r];
    *(v8bf*)(vT + (size_t)cc * B_TOK + m0) = pk;
  }
}

__global__ __launch_bounds__(256)
void qkv_gemm_kernel(const bf16* __restrict__ A, const bf16* __restrict__ Bt,
                     bf16* __restrict__ q, bf16* __restrict__ k,
                     bf16* __restrict__ vT) {
  const int wave = threadIdx.x >> 5;
  const int m_wave = blockIdx.x * 64 + (wave & 1) * 32;
  const int n_wave = blockIdx.y * 128 + (wave >> 1) * 32;

  v8f c00 = zero_v8f(), c01 = zero_v8f(), c10 = zero_v8f(), c11 = zero_v8f();
#pragma unroll 4
  for (int kc = 0; kc < DIM; kc += 32) {
    v16bf a0 = load_frag_a(A + (size_t)m_wave * DIM + kc, DIM);
    v16bf a1 = load_frag_a(A + (size_t)(m_wave + 16) * DIM + kc, DIM);
    v16bf b0 = load_frag_b(Bt + (size_t)n_wave * DIM + kc, DIM);
    v16bf b1 = load_frag_b(Bt + (size_t)(n_wave + 16) * DIM + kc, DIM);
    c00 = WMMA_BF16(a0, b0, c00);
    c01 = WMMA_BF16(a0, b1, c01);
    c10 = WMMA_BF16(a1, b0, c10);
    c11 = WMMA_BF16(a1, b1, c11);
  }
  store_qkv_tile(c00, m_wave,      n_wave,      q, k, vT);
  store_qkv_tile(c01, m_wave,      n_wave + 16, q, k, vT);
  store_qkv_tile(c10, m_wave + 16, n_wave,      q, k, vT);
  store_qkv_tile(c11, m_wave + 16, n_wave + 16, q, k, vT);
}

// ---------------------------------------------------------------------------
// Flash attention with post-softmax label mask.
// One wave owns (head h, 16 query rows); streams j in tiles of 32.
// ---------------------------------------------------------------------------
__global__ __launch_bounds__(256)
void attn_kernel(const bf16* __restrict__ q, const bf16* __restrict__ k,
                 const bf16* __restrict__ vT, const int* __restrict__ labels,
                 bf16* __restrict__ attn_out) {
  // per-wave P-transpose tile; 40-element (80 B) row stride -> conflict-free
  __shared__ bf16 pbuf[8][16][40];
  const float scale = 0.1020620726159657f;  // 96^-0.5

  const int wave = threadIdx.x >> 5;
  const int lane = threadIdx.x & 31;
  const int half = lane >> 4;
  const int col  = lane & 15;
  const int gw = blockIdx.x * 8 + wave;     // 2048 waves total
  const int h = gw >> 8;                    // 256 i-tiles per head
  const int i_base = (gw & 255) << 4;

  // Q fragments for this wave's 16 rows (K = 96 -> 3 chunks), reused all j.
  const bf16* qbase = q + (size_t)i_base * INNER + h * DH;
  v16bf qf0 = load_frag_a(qbase,      INNER);
  v16bf qf1 = load_frag_a(qbase + 32, INNER);
  v16bf qf2 = load_frag_a(qbase + 64, INNER);

  int li[8];                                // labels of my 8 rows (per half)
#pragma unroll
  for (int r = 0; r < 8; ++r) li[r] = labels[i_base + half * 8 + r];

  float mrow[8], lrow[8];
#pragma unroll
  for (int r = 0; r < 8; ++r) { mrow[r] = -3.0e38f; lrow[r] = 0.0f; }
  v8f acc[6];
#pragma unroll
  for (int t = 0; t < 6; ++t) acc[t] = zero_v8f();

  for (int jb = 0; jb < B_TOK; jb += 32) {
    // S = q . k^T  (two 16-wide j sub-tiles, K=96)
    const bf16* kb0 = k + (size_t)jb * INNER + h * DH;
    const bf16* kb1 = kb0 + (size_t)16 * INNER;
    v8f s0 = zero_v8f(), s1 = zero_v8f();
    s0 = WMMA_BF16(qf0, load_frag_b(kb0,      INNER), s0);
    s1 = WMMA_BF16(qf0, load_frag_b(kb1,      INNER), s1);
    s0 = WMMA_BF16(qf1, load_frag_b(kb0 + 32, INNER), s0);
    s1 = WMMA_BF16(qf1, load_frag_b(kb1 + 32, INNER), s1);
    s0 = WMMA_BF16(qf2, load_frag_b(kb0 + 64, INNER), s0);
    s1 = WMMA_BF16(qf2, load_frag_b(kb1 + 64, INNER), s1);

    const int labj0 = labels[jb + col];
    const int labj1 = labels[jb + 16 + col];

    // Streaming softmax. C-layout: vgpr r / lane -> row (r + 8*half), so row
    // reductions are 16-lane butterflies (masks 1,2,4,8 stay inside a half).
    float p0[8], p1[8];
#pragma unroll
    for (int r = 0; r < 8; ++r) {
      float v0 = s0[r] * scale;
      float v1 = s1[r] * scale;
      float mx = fmaxf(v0, v1);
#pragma unroll
      for (int off = 1; off < 16; off <<= 1)
        mx = fmaxf(mx, __shfl_xor(mx, off, 32));
      float mnew  = fmaxf(mrow[r], mx);
      float alpha = __expf(mrow[r] - mnew);
      float e0 = __expf(v0 - mnew);
      float e1 = __expf(v1 - mnew);
      float rs = e0 + e1;
#pragma unroll
      for (int off = 1; off < 16; off <<= 1)
        rs += __shfl_xor(rs, off, 32);
      lrow[r] = lrow[r] * alpha + rs;       // denominator: UNmasked sum
      mrow[r] = mnew;
#pragma unroll
      for (int t = 0; t < 6; ++t) acc[t][r] *= alpha;
      p0[r] = (li[r] == labj0) ? e0 : 0.0f; // numerator: masked
      p1[r] = (li[r] == labj1) ? e1 : 0.0f;
    }

    // Transpose P (C-layout -> A-layout) through LDS.
#pragma unroll
    for (int r = 0; r < 8; ++r) {
      pbuf[wave][half * 8 + r][col]      = (bf16)p0[r];
      pbuf[wave][half * 8 + r][16 + col] = (bf16)p1[r];
    }
    asm volatile("s_wait_dscnt 0" ::: "memory");  // CDNA5 split DS counter

    v16bf pa;
    {
      const bf16* prow = &pbuf[wave][col][half << 3];
      v8bf lo = *(const v8bf*)(prow);
      v8bf hi = *(const v8bf*)(prow + 16);
#pragma unroll
      for (int i = 0; i < 8; ++i) { pa[i] = lo[i]; pa[i + 8] = hi[i]; }
    }

    // acc += P . V  (B fragments from pre-transposed vT, contiguous in j)
    const bf16* vbase = vT + (size_t)(h * DH) * B_TOK + jb;
#pragma unroll
    for (int t = 0; t < 6; ++t) {
      v16bf bv = load_frag_b(vbase + (size_t)(t * 16) * B_TOK, B_TOK);
      acc[t] = WMMA_BF16(pa, bv, acc[t]);
    }
  }

  // out = acc / l, laid out 'h i d -> i (h d)' as bf16 A-matrix for out-proj
#pragma unroll
  for (int r = 0; r < 8; ++r) lrow[r] = 1.0f / lrow[r];
  bf16* obase = attn_out + (size_t)(i_base + half * 8) * INNER + h * DH + col;
#pragma unroll
  for (int t = 0; t < 6; ++t)
#pragma unroll
    for (int r = 0; r < 8; ++r)
      obase[(size_t)r * INNER + t * 16] = (bf16)(acc[t][r] * lrow[r]);
}

// ---------------------------------------------------------------------------
// Output projection: [4096,768] x [768,768] -> f32 d_out
// ---------------------------------------------------------------------------
__global__ __launch_bounds__(256)
void out_gemm_kernel(const bf16* __restrict__ A, const bf16* __restrict__ Bt,
                     float* __restrict__ out) {
  const int wave = threadIdx.x >> 5;
  const int lane = threadIdx.x & 31;
  const int m_wave = blockIdx.x * 64 + (wave & 1) * 32;
  const int n_wave = blockIdx.y * 128 + (wave >> 1) * 32;

  v8f c00 = zero_v8f(), c01 = zero_v8f(), c10 = zero_v8f(), c11 = zero_v8f();
#pragma unroll 4
  for (int kc = 0; kc < INNER; kc += 32) {
    v16bf a0 = load_frag_a(A + (size_t)m_wave * INNER + kc, INNER);
    v16bf a1 = load_frag_a(A + (size_t)(m_wave + 16) * INNER + kc, INNER);
    v16bf b0 = load_frag_b(Bt + (size_t)n_wave * INNER + kc, INNER);
    v16bf b1 = load_frag_b(Bt + (size_t)(n_wave + 16) * INNER + kc, INNER);
    c00 = WMMA_BF16(a0, b0, c00);
    c01 = WMMA_BF16(a0, b1, c01);
    c10 = WMMA_BF16(a1, b0, c10);
    c11 = WMMA_BF16(a1, b1, c11);
  }
  const int ncol = lane & 15;
  const int m0 = (lane >> 4) << 3;
  const v8f cs[4] = {c00, c01, c10, c11};
#pragma unroll
  for (int t = 0; t < 4; ++t) {
    const int mt = m_wave + (t >> 1) * 16 + m0;
    const int nt = n_wave + (t & 1) * 16 + ncol;
#pragma unroll
    for (int r = 0; r < 8; ++r) out[(size_t)(mt + r) * DIM + nt] = cs[t][r];
  }
}

// ---------------------------------------------------------------------------
// Host launcher
// ---------------------------------------------------------------------------
extern "C" void kernel_launch(void* const* d_in, const int* in_sizes, int n_in,
                              void* d_out, int out_size, void* d_ws, size_t ws_size,
                              hipStream_t stream) {
  (void)in_sizes; (void)n_in; (void)out_size; (void)ws_size;
  const float* x      = (const float*)d_in[0];
  const int*   labels = (const int*)d_in[1];
  const float* Wqkv   = (const float*)d_in[2];
  const float* Wout   = (const float*)d_in[3];
  float*       out    = (float*)d_out;

  char* ws = (char*)d_ws;
  size_t off = 0;
  auto take = [&](size_t bytes) -> char* {
    char* p = ws + off;
    off += (bytes + 255) & ~(size_t)255;
    return p;
  };
  bf16* x_bf  = (bf16*)take((size_t)B_TOK * DIM  * sizeof(bf16));
  bf16* wqkvT = (bf16*)take((size_t)NQKV  * DIM  * sizeof(bf16));
  bf16* woutT = (bf16*)take((size_t)DIM   * INNER * sizeof(bf16));
  bf16* qm    = (bf16*)take((size_t)B_TOK * INNER * sizeof(bf16));
  bf16* km    = (bf16*)take((size_t)B_TOK * INNER * sizeof(bf16));
  bf16* vTm   = (bf16*)take((size_t)INNER * B_TOK * sizeof(bf16));
  bf16* attn  = (bf16*)take((size_t)B_TOK * INNER * sizeof(bf16));

  {
    int n = B_TOK * DIM;
    cvt_f32_bf16_kernel<<<(n + 255) / 256, 256, 0, stream>>>(x, x_bf, n);
  }
  transpose_f32_bf16_kernel<<<(DIM * NQKV + 255) / 256, 256, 0, stream>>>(
      Wqkv, wqkvT, DIM, NQKV);                 // [768,2304] -> [2304,768]
  transpose_f32_bf16_kernel<<<(INNER * DIM + 255) / 256, 256, 0, stream>>>(
      Wout, woutT, INNER, DIM);                // [768,768]  -> [768,768]^T

  dim3 g1(B_TOK / 64, NQKV / 128);             // 64 x 18
  qkv_gemm_kernel<<<g1, 256, 0, stream>>>(x_bf, wqkvT, qm, km, vTm);

  attn_kernel<<<(HEADS * (B_TOK / 16)) / 8, 256, 0, stream>>>(
      qm, km, vTm, labels, attn);              // 256 blocks x 8 waves

  dim3 g2(B_TOK / 64, DIM / 128);              // 64 x 6
  out_gemm_kernel<<<g2, 256, 0, stream>>>(attn, woutT, out);
}
